// LinearAttention_74526272520399
// MI455X (gfx1250) — matrix operand order repaired and checked by reading, not compiled, not run on hardware
//
#include <hip/hip_runtime.h>
#include <hip/hip_bf16.h>

// ---------------------------------------------------------------------------
// Linear attention, MI455X (gfx1250, wave32, WMMA).
// All heavy math runs through v_wmma_f32_16x16x32_bf16 (bf16 in, f32 accum).
// Pipeline:
//   1) convw      : Wq/Wk/Wv/Wo f32 -> bf16
//   2) proj<0..2> : q/k/v = act(X @ W^T + b); K-mode fuses mask + per-head row L2 norm
//   3) colsumsq   : sum over n of q_feat^2 per (b, feature)  (q norm is over seq axis)
//   4) kv_kernel  : per (b,h) kv = k^T v   (64x64, K=4096)   via WMMA
//   5) qkv_out    : per (b,h) out = q_norm @ kv               via WMMA
//   6) proj<3>    : d_out = attn @ Wo^T + bo (f32 out)        via WMMA
// B tiles (and bf16 A tiles in the final GEMM) are staged with
// GLOBAL_LOAD_ASYNC_TO_LDS_B128 (ASYNCcnt-tracked DMA into LDS), with a
// plain uint4 copy fallback if the builtin is not exposed.
// Workspace layout (needs ~267 MB):
//   [0,8MB)      4x bf16 weights
//   [8,72MB)     q_feat bf16   [72,136MB) k_feat bf16   [136,200MB) v_proj bf16
//   [200,264MB)  attn bf16     [264MB)    sumsq f32 (32KB)  [265MB) kv f32 (2MB)
// ---------------------------------------------------------------------------

typedef __attribute__((ext_vector_type(16))) __bf16 v16bf;
typedef __attribute__((ext_vector_type(8)))  float  v8f;

#define D      1024
#define NSEQ   4096
#define NHEAD  16
#define DH     64
#define BM     128
#define BN     64
#define BK     32
#define LDSAB  (BK + 8)   // 40 halves: rows start 16B-aligned, no bank conflicts
#define LDKV   (DH + 8)   // 72 halves

#if defined(__gfx1250__) && __has_builtin(__builtin_amdgcn_global_load_async_to_lds_b128)
#define HAVE_ASYNC_LDS 1
#else
#define HAVE_ASYNC_LDS 0
#endif

#if HAVE_ASYNC_LDS
typedef int v4i_vec __attribute__((vector_size(16)));          // matches builtin param
typedef __attribute__((address_space(1))) v4i_vec* gv4i_p;     // global int4*
typedef __attribute__((address_space(3))) v4i_vec* lv4i_p;     // LDS int4*
#endif

// 16B global -> LDS copy; async DMA on gfx1250 when available.
__device__ __forceinline__ void g2lds_b128(const void* g, void* l) {
#if HAVE_ASYNC_LDS
    __builtin_amdgcn_global_load_async_to_lds_b128(
        (gv4i_p)(__attribute__((address_space(1))) const void*)(g),
        (lv4i_p)(__attribute__((address_space(3))) void*)(l),
        0, 0);
#else
    *(uint4*)l = *(const uint4*)g;
#endif
}
__device__ __forceinline__ void async_wait() {
#if HAVE_ASYNC_LDS
#if __has_builtin(__builtin_amdgcn_s_wait_asynccnt)
    __builtin_amdgcn_s_wait_asynccnt(0);
#else
    asm volatile("s_wait_asynccnt 0" ::: "memory");
#endif
#endif
}

__device__ __forceinline__ unsigned short f2bf(float x) {
    unsigned int u = __float_as_uint(x);
    return (unsigned short)((u + 0x7FFFu + ((u >> 16) & 1u)) >> 16); // RNE
}
__device__ __forceinline__ float bf2f(unsigned short h) {
    return __uint_as_float(((unsigned int)h) << 16);
}
__device__ __forceinline__ float elu1(float x) {           // 1 + elu(x)
    return x > 0.f ? x + 1.f : __expf(x);
}

union FragU { v16bf v; unsigned int u[8]; };

// A fragment (16x32 bf16): lane = (m = lane&15, half = lane>>4).
// elements 0..7  -> K = half*8 + e      (contiguous 16B)
// elements 8..15 -> K = 16 + half*8 + e (contiguous 16B)
__device__ __forceinline__ v16bf frag_A(const unsigned short* base, int stride,
                                        int m, int hf) {
    FragU f;
    const unsigned short* p = base + m * stride;
    uint4 a0 = *(const uint4*)(p + hf * 8);
    uint4 a1 = *(const uint4*)(p + 16 + hf * 8);
    f.u[0] = a0.x; f.u[1] = a0.y; f.u[2] = a0.z; f.u[3] = a0.w;
    f.u[4] = a1.x; f.u[5] = a1.y; f.u[6] = a1.z; f.u[7] = a1.w;
    return f.v;
}
// B fragment (32x16 bf16) from a transposed [n][k] LDS tile:
// lane = (n = lane&15, half = lane>>4), elements 0..7 -> K = k0 + half*8 + e, 8..15 -> K = k0 + 16 + half*8 + (e-8).
__device__ __forceinline__ v16bf frag_BT(const unsigned short* base, int stride,
                                         int n, int hf, int k0) {
    FragU f;
    const unsigned short* p = base + n * stride + k0 + hf * 8;
    uint4 a0 = *(const uint4*)(p);
    uint4 a1 = *(const uint4*)(p + 16);
    f.u[0] = a0.x; f.u[1] = a0.y; f.u[2] = a0.z; f.u[3] = a0.w;
    f.u[4] = a1.x; f.u[5] = a1.y; f.u[6] = a1.z; f.u[7] = a1.w;
    return f.v;
}
#define WMMA_BF16(a, b, c) \
    __builtin_amdgcn_wmma_f32_16x16x32_bf16(false, (a), false, (b), (short)0, (c), false, false)

// --------------------------- weight f32 -> bf16 ----------------------------
__global__ void convw_kernel(const float* __restrict__ w, unsigned short* __restrict__ o) {
    int idx = (blockIdx.x * 256 + threadIdx.x) * 4;
    float4 f = *(const float4*)(w + idx);
    alignas(8) unsigned short h[4] = { f2bf(f.x), f2bf(f.y), f2bf(f.z), f2bf(f.w) };
    *(uint2*)(o + idx) = *(const uint2*)h;
}

// ------------------- projection GEMM: out = act(X @ W^T + b) ---------------
// MODE 0: Q (elu+1, bf16 out)   MODE 1: K (elu+1, mask, per-head row L2 norm)
// MODE 2: V (bf16 out)          MODE 3: final (bf16 A input, f32 out)
template<int MODE>
__global__ __launch_bounds__(256)
void proj_kernel(const float* __restrict__ X, const unsigned short* __restrict__ Xbf,
                 const unsigned short* __restrict__ Wbf, const float* __restrict__ bias,
                 const unsigned char* __restrict__ mask,
                 unsigned short* __restrict__ outb, float* __restrict__ outf) {
    __shared__ unsigned short sA[BM][LDSAB];
    __shared__ unsigned short sB[BN][LDSAB];   // sB[n][k] = W[n0+n][k0+k] (= B^T)

    const int tid  = threadIdx.x;
    const int wave = tid >> 5;
    const int lane = tid & 31;
    const int hf   = lane >> 4;
    const int lnn  = lane & 15;
    const int m0 = blockIdx.y * BM;
    const int n0 = blockIdx.x * BN;

    v8f zero = {0.f, 0.f, 0.f, 0.f, 0.f, 0.f, 0.f, 0.f};
    v8f acc[4];
#pragma unroll
    for (int t = 0; t < 4; ++t) acc[t] = zero;

    const int arow = tid >> 1, aseg = tid & 1;   // A staging: 128 rows x 2 x 16 elems
    const int brow = tid >> 2, bch  = tid & 3;   // B staging: 64 rows x 4 x 8 elems

    for (int k0 = 0; k0 < D; k0 += BK) {
        // stage B tile (bf16 weights) via async global->LDS DMA
        g2lds_b128(Wbf + (size_t)(n0 + brow) * D + k0 + bch * 8, &sB[brow][bch * 8]);

        if constexpr (MODE == 3) {
            const unsigned short* src = Xbf + (size_t)(m0 + arow) * D + k0 + aseg * 16;
            g2lds_b128(src,     &sA[arow][aseg * 16]);
            g2lds_b128(src + 8, &sA[arow][aseg * 16 + 8]);
        } else {
            const float* src = X + (size_t)(m0 + arow) * D + k0 + aseg * 16;
            alignas(16) float fv[16];
            *(float4*)&fv[0]  = *(const float4*)(src);
            *(float4*)&fv[4]  = *(const float4*)(src + 4);
            *(float4*)&fv[8]  = *(const float4*)(src + 8);
            *(float4*)&fv[12] = *(const float4*)(src + 12);
            alignas(16) unsigned short hv[16];
#pragma unroll
            for (int i = 0; i < 16; ++i) hv[i] = f2bf(fv[i]);
            *(uint4*)&sA[arow][aseg * 16]     = *(const uint4*)&hv[0];
            *(uint4*)&sA[arow][aseg * 16 + 8] = *(const uint4*)&hv[8];
        }

        if (k0 + BK < D)   // gfx1250 global_prefetch_b8 for the next weight tile
            __builtin_prefetch(Wbf + (size_t)(n0 + brow) * D + k0 + BK + bch * 8, 0, 1);

        async_wait();      // drain ASYNCcnt before the barrier makes LDS visible
        __syncthreads();
        v16bf a = frag_A(&sA[wave * 16][0], LDSAB, lnn, hf);
#pragma unroll
        for (int t = 0; t < 4; ++t) {
            v16bf bfr = frag_BT(&sB[t * 16][0], LDSAB, lnn, hf, 0);
            acc[t] = WMMA_BF16(a, bfr, acc[t]);
        }
        __syncthreads();
    }

    if constexpr (MODE == 1) {
        // epilogue: bias -> elu+1 -> zero padded keys -> per-row (per-head) L2 norm
        __shared__ float sC[BM][BN + 1];
        __shared__ float rs[BM];
#pragma unroll
        for (int t = 0; t < 4; ++t) {
#pragma unroll
            for (int r = 0; r < 8; ++r) {
                int lm = wave * 16 + hf * 8 + r;
                int gm = m0 + lm;                       // flat (b*NSEQ + n) row
                int gn = n0 + t * 16 + lnn;
                float val = elu1(acc[t][r] + bias[gn]);
                if (mask[gm]) val = 0.f;
                sC[lm][t * 16 + lnn] = val;
            }
        }
        __syncthreads();
        if (tid < BM) {
            float s = 0.f;
#pragma unroll 8
            for (int c = 0; c < BN; ++c) { float v = sC[tid][c]; s += v * v; }
            rs[tid] = 1.f / fmaxf(sqrtf(s), 1e-12f);
        }
        __syncthreads();
        {
            int row = tid >> 1, c0 = (tid & 1) * 32;
            float sc = rs[row];
#pragma unroll 8
            for (int c = 0; c < 32; ++c)
                outb[(size_t)(m0 + row) * D + n0 + c0 + c] = f2bf(sC[row][c0 + c] * sc);
        }
    } else {
#pragma unroll
        for (int t = 0; t < 4; ++t) {
#pragma unroll
            for (int r = 0; r < 8; ++r) {
                int gm = m0 + wave * 16 + hf * 8 + r;
                int gn = n0 + t * 16 + lnn;
                float val = acc[t][r] + bias[gn];
                if constexpr (MODE == 0) val = elu1(val);
                if constexpr (MODE == 3) outf[(size_t)gm * D + gn] = val;
                else                     outb[(size_t)gm * D + gn] = f2bf(val);
            }
        }
    }
}

// ------------- q column sum-of-squares: q is normalized over seq axis ------
__global__ void colsumsq_kernel(const unsigned short* __restrict__ qf,
                                float* __restrict__ ss) {
    int idx = blockIdx.x * 256 + threadIdx.x;   // 8 * 1024 = 8192
    int b = idx >> 10, c = idx & (D - 1);
    const unsigned short* p = qf + (size_t)b * NSEQ * D + c;
    float s = 0.f;
    for (int n = 0; n < NSEQ; ++n) { float v = bf2f(p[(size_t)n * D]); s += v * v; }
    ss[idx] = s;
}

// ---------------- per (b,h): kv[64][64] = k^T @ v over n=4096 --------------
__global__ __launch_bounds__(256)
void kv_kernel(const unsigned short* __restrict__ kf, const unsigned short* __restrict__ vf,
               float* __restrict__ kv) {
    __shared__ unsigned short sK[DH][LDSAB];   // sK[d][n]  (k transposed)
    __shared__ unsigned short sV[DH][LDSAB];   // sV[e][n]  (v transposed = B^T)
    const int tid  = threadIdx.x;
    const int wave = tid >> 5, lane = tid & 31, hf = lane >> 4, lnn = lane & 15;
    const int bh = blockIdx.x, b = bh >> 4, h = bh & 15;
    const int mtile = wave >> 1;          // 8 waves x 2 tiles = 4x4 output tiles
    const int nt0   = (wave & 1) * 2;
    v8f z = {0.f, 0.f, 0.f, 0.f, 0.f, 0.f, 0.f, 0.f};
    v8f acc0 = z, acc1 = z;

    const int si = tid & 31, sj = (tid >> 5) * 8;
    for (int nn = 0; nn < NSEQ; nn += 32) {
        size_t rowoff = ((size_t)b * NSEQ + nn + si) * D + (size_t)h * DH + sj;
        uint4 ku = *(const uint4*)(kf + rowoff);
        uint4 vu = *(const uint4*)(vf + rowoff);
        alignas(16) unsigned short kh[8]; *(uint4*)kh = ku;
        alignas(16) unsigned short vh[8]; *(uint4*)vh = vu;
#pragma unroll
        for (int j = 0; j < 8; ++j) { sK[sj + j][si] = kh[j]; sV[sj + j][si] = vh[j]; }
        __syncthreads();
        v16bf a  = frag_A (&sK[mtile * 16][0],     LDSAB, lnn, hf);
        v16bf b0 = frag_BT(&sV[nt0 * 16][0],       LDSAB, lnn, hf, 0);
        v16bf b1 = frag_BT(&sV[(nt0 + 1) * 16][0], LDSAB, lnn, hf, 0);
        acc0 = WMMA_BF16(a, b0, acc0);
        acc1 = WMMA_BF16(a, b1, acc1);
        __syncthreads();
    }
#pragma unroll
    for (int r = 0; r < 8; ++r) {
        int m = mtile * 16 + hf * 8 + r;
        kv[((size_t)bh * DH + m) * DH + nt0 * 16 + lnn]       = acc0[r];
        kv[((size_t)bh * DH + m) * DH + (nt0 + 1) * 16 + lnn] = acc1[r];
    }
}

// ------------- per (b,h): attn = (q / max(colnorm,eps)) @ kv ---------------
__global__ __launch_bounds__(256)
void qkv_out_kernel(const unsigned short* __restrict__ qf, const float* __restrict__ ss,
                    const float* __restrict__ kv, unsigned short* __restrict__ attn) {
    __shared__ unsigned short sQ[BM][LDSAB];
    __shared__ unsigned short sKV[DH][LDKV];   // sKV[e][k] = kv[k][e] bf16 (B^T)
    const int tid  = threadIdx.x;
    const int wave = tid >> 5, lane = tid & 31, hf = lane >> 4, lnn = lane & 15;
    const int bh = blockIdx.y, b = bh >> 4, h = bh & 15;
    const int m0 = blockIdx.x * BM;

    {   // stage kv transposed, f32 -> bf16 (once; covered by first in-loop barrier)
        int e = tid & 63, kg = (tid >> 6) * 16;
#pragma unroll
        for (int k = 0; k < 16; ++k)
            sKV[e][kg + k] = f2bf(kv[((size_t)bh * DH + kg + k) * DH + e]);
    }

    v8f zero = {0.f, 0.f, 0.f, 0.f, 0.f, 0.f, 0.f, 0.f};
    v8f acc[4];
#pragma unroll
    for (int t = 0; t < 4; ++t) acc[t] = zero;
    const int arow = tid >> 1, aseg = tid & 1;

#pragma unroll
    for (int ks = 0; ks < 2; ++ks) {
        int col0 = h * DH + ks * 32 + aseg * 16;
        const unsigned short* src = qf + ((size_t)b * NSEQ + m0 + arow) * D + col0;
        uint4 u0 = *(const uint4*)src;
        uint4 u1 = *(const uint4*)(src + 8);
        alignas(16) unsigned short hv[16];
        *(uint4*)&hv[0] = u0; *(uint4*)&hv[8] = u1;
        const float* sp = ss + b * D + col0;
        alignas(16) unsigned short ov[16];
#pragma unroll
        for (int i = 0; i < 16; ++i) {
            float sc = 1.f / fmaxf(sqrtf(sp[i]), 1e-12f);
            ov[i] = f2bf(bf2f(hv[i]) * sc);
        }
        *(uint4*)&sQ[arow][aseg * 16]     = *(const uint4*)&ov[0];
        *(uint4*)&sQ[arow][aseg * 16 + 8] = *(const uint4*)&ov[8];
        __syncthreads();
        v16bf a = frag_A(&sQ[wave * 16][0], LDSAB, lnn, hf);
#pragma unroll
        for (int t = 0; t < 4; ++t) {
            v16bf bb = frag_BT(&sKV[t * 16][0], LDKV, lnn, hf, ks * 32);
            acc[t] = WMMA_BF16(a, bb, acc[t]);
        }
        __syncthreads();
    }
#pragma unroll
    for (int t = 0; t < 4; ++t) {
#pragma unroll
        for (int r = 0; r < 8; ++r) {
            int gm = m0 + wave * 16 + hf * 8 + r;
            attn[((size_t)b * NSEQ + gm) * D + h * DH + t * 16 + lnn] = f2bf(acc[t][r]);
        }
    }
}

// ---------------------------------------------------------------------------
extern "C" void kernel_launch(void* const* d_in, const int* in_sizes, int n_in,
                              void* d_out, int out_size, void* d_ws, size_t ws_size,
                              hipStream_t stream) {
    (void)in_sizes; (void)n_in; (void)out_size; (void)ws_size;
    const float* query = (const float*)d_in[0];
    const float* key   = (const float*)d_in[1];
    const float* value = (const float*)d_in[2];
    const unsigned char* mask = (const unsigned char*)d_in[3];   // bool, 1 byte/elem
    const float* Wq = (const float*)d_in[4];  const float* bq = (const float*)d_in[5];
    const float* Wk = (const float*)d_in[6];  const float* bk = (const float*)d_in[7];
    const float* Wv = (const float*)d_in[8];  const float* bv = (const float*)d_in[9];
    const float* Wo = (const float*)d_in[10]; const float* bo = (const float*)d_in[11];
    float* out = (float*)d_out;

    char* ws = (char*)d_ws;
    const size_t MB = 1024u * 1024u;
    unsigned short* wq_bf  = (unsigned short*)(ws + 0 * MB);
    unsigned short* wk_bf  = (unsigned short*)(ws + 2 * MB);
    unsigned short* wv_bf  = (unsigned short*)(ws + 4 * MB);
    unsigned short* wo_bf  = (unsigned short*)(ws + 6 * MB);
    unsigned short* q_feat = (unsigned short*)(ws + 8 * MB);
    unsigned short* k_feat = (unsigned short*)(ws + 72 * MB);
    unsigned short* v_proj = (unsigned short*)(ws + 136 * MB);
    unsigned short* attn   = (unsigned short*)(ws + 200 * MB);
    float* sumsq = (float*)(ws + 264 * MB);
    float* kvbuf = (float*)(ws + 265 * MB);

    dim3 blk(256);
    convw_kernel<<<1024, blk, 0, stream>>>(Wq, wq_bf);
    convw_kernel<<<1024, blk, 0, stream>>>(Wk, wk_bf);
    convw_kernel<<<1024, blk, 0, stream>>>(Wv, wv_bf);
    convw_kernel<<<1024, blk, 0, stream>>>(Wo, wo_bf);

    dim3 g1(D / BN, (8 * NSEQ) / BM);   // (16, 256)
    proj_kernel<0><<<g1, blk, 0, stream>>>(query, nullptr, wq_bf, bq, nullptr, q_feat, nullptr);
    proj_kernel<1><<<g1, blk, 0, stream>>>(key,   nullptr, wk_bf, bk, mask,    k_feat, nullptr);
    proj_kernel<2><<<g1, blk, 0, stream>>>(value, nullptr, wv_bf, bv, nullptr, v_proj, nullptr);

    colsumsq_kernel<<<32, blk, 0, stream>>>(q_feat, sumsq);
    kv_kernel<<<8 * NHEAD, blk, 0, stream>>>(k_feat, v_proj, kvbuf);
    qkv_out_kernel<<<dim3(NSEQ / BM, 8 * NHEAD), blk, 0, stream>>>(q_feat, sumsq, kvbuf, attn);
    proj_kernel<3><<<g1, blk, 0, stream>>>(nullptr, attn, wo_bf, bo, nullptr, nullptr, out);
}
